// GAT_13134009991665
// MI455X (gfx1250) — compile-verified, matrix operand order checked by reading, not run on hardware
//
#include <hip/hip_runtime.h>
#include <hip/hip_bf16.h>

typedef __attribute__((ext_vector_type(16))) __bf16 v16bf;
typedef __attribute__((ext_vector_type(8)))  float  v8f;

#define D 128

// ---------------- helpers ----------------
__device__ __forceinline__ unsigned f32_ord(float f) {
  unsigned b = __float_as_uint(f);
  return (b >> 31) ? ~b : (b | 0x80000000u);   // monotonic float->uint
}
__device__ __forceinline__ float ord_f32(unsigned u) {
  return (u >> 31) ? __uint_as_float(u & 0x7fffffffu) : __uint_as_float(~u);
}

// ---------------- conversion / packing ----------------
__global__ void cvt_f32_bf16(const float* __restrict__ in, __bf16* __restrict__ out, int n) {
  int i = blockIdx.x * blockDim.x + threadIdx.x;
  if (i < n) out[i] = (__bf16)in[i];
}

// Pack row-major W[k*128+n] into WMMA B-fragment order:
// WP[((kk*8+nt)*32 + lane)*16 + i] = W[(kk*32 + (lane>>4)*16 + i)*128 + nt*16 + (lane&15)]
// so each lane's 16-element fragment is a contiguous 32B load.
__global__ void pack_w(const float* __restrict__ W, __bf16* __restrict__ WP) {
  int idx = blockIdx.x * blockDim.x + threadIdx.x;
  if (idx >= D * D) return;
  int i    = idx & 15;
  int lane = (idx >> 4) & 31;
  int nt   = (idx >> 9) & 7;
  int kk   = (idx >> 12) & 3;
  int n = nt * 16 + (lane & 15);
  int k = kk * 32 + (lane >> 4) * 16 + i;
  WP[idx] = (__bf16)W[k * D + n];
}

// ---------------- WMMA GEMM: out[N,128] = X[N,128] @ W[128,128] ----------------
// W (pre-packed in fragment order) is staged into LDS once per workgroup via the
// CDNA5 async-copy path, then WMMA feeds from ds_load_b128.
// Each wave computes a 32-row x 128-col tile: 2 A-fragments per k-step, 16 v8f accumulators.
__global__ void __launch_bounds__(256) gemm_bf16(
    const __bf16* __restrict__ X,
    const __bf16* __restrict__ WPl, const __bf16* __restrict__ WPr,
    float* __restrict__ outL, float* __restrict__ outR, int nrows) {
  __shared__ __bf16 sW[D * D];   // 32 KB packed weight

  const __bf16* WP = blockIdx.y ? WPr : WPl;
  float*       out = blockIdx.y ? outR : outL;

  // ---- async stage: global -> LDS, 128 B per thread (8 x b128 per lane) ----
  {
    unsigned lds_base = (unsigned)(uintptr_t)(&sW[0]);
    unsigned long long g_base = (unsigned long long)(uintptr_t)WP;
#pragma unroll
    for (int j = 0; j < 8; ++j) {
      unsigned byteOff = (threadIdx.x + j * 256) * 16;
      unsigned lds_addr = lds_base + byteOff;
      unsigned long long g_addr = g_base + byteOff;
      asm volatile("global_load_async_to_lds_b128 %0, %1, off"
                   :: "v"(lds_addr), "v"(g_addr) : "memory");
    }
    asm volatile("s_wait_asynccnt 0x0" ::: "memory");
    __syncthreads();   // all threads reach this (no earlier returns)
  }

  int wave = threadIdx.x >> 5;
  int lane = threadIdx.x & 31;
  int pair = blockIdx.x * 8 + wave;     // 32 rows per wave
  int row0 = pair * 32;
  if (row0 >= nrows) return;

  int m  = lane & 15;
  int kh = lane >> 4;
  int rA = row0 + m;       if (rA >= nrows) rA = nrows - 1;  // clamp loads; stores guarded
  int rB = row0 + 16 + m;  if (rB >= nrows) rB = nrows - 1;

  v8f c0[8] = {};
  v8f c1[8] = {};
#pragma unroll
  for (int kk = 0; kk < 4; ++kk) {
    v16bf a0 = *(const v16bf*)(X + (size_t)rA * D + kk * 32 + kh * 16);
    v16bf a1 = *(const v16bf*)(X + (size_t)rB * D + kk * 32 + kh * 16);
#pragma unroll
    for (int nt = 0; nt < 8; ++nt) {
      v16bf b = *(const v16bf*)(sW + (size_t)(((kk * 8) + nt) * 32 + lane) * 16);
      c0[nt] = __builtin_amdgcn_wmma_f32_16x16x32_bf16(
          false, a0, false, b, (short)0, c0[nt], false, false);
      c1[nt] = __builtin_amdgcn_wmma_f32_16x16x32_bf16(
          false, a1, false, b, (short)0, c1[nt], false, false);
    }
  }

  // C/D layout: VGPR i -> M = i (lanes 0-15) or M = 8+i (lanes 16-31), N = lane&15
  int col0 = lane & 15;
  int mb0  = row0 + (kh ? 8 : 0);
  int mb1  = row0 + 16 + (kh ? 8 : 0);
#pragma unroll
  for (int nt = 0; nt < 8; ++nt) {
#pragma unroll
    for (int i = 0; i < 8; ++i) {
      int r0 = mb0 + i;
      if (r0 < nrows) out[(size_t)r0 * D + nt * 16 + col0] = c0[nt][i];
      int r1 = mb1 + i;
      if (r1 < nrows) out[(size_t)r1 * D + nt * 16 + col0] = c1[nt][i];
    }
  }
}

// ---------------- edge phase ----------------
// Pass A: e[k] = att . leaky_relu(xl[src]+xr[dst]); segment max via ordered-uint atomicMax
__global__ void __launch_bounds__(256) edge_score(
    const float* __restrict__ xl, const float* __restrict__ xr,
    const int* __restrict__ src, const int* __restrict__ dst,
    const float* __restrict__ att, float* __restrict__ e,
    unsigned* __restrict__ menc, int E) {
  int lane = threadIdx.x & 31;
  int k = blockIdx.x * 8 + (threadIdx.x >> 5);
  if (k >= E) return;
  int s = src[k], d = dst[k];
  float4 vl = ((const float4*)(xl + (size_t)s * D))[lane];
  float4 vr = ((const float4*)(xr + (size_t)d * D))[lane];
  float4 va = ((const float4*)att)[lane];
  float zx = vl.x + vr.x; zx = zx > 0.f ? zx : 0.2f * zx;
  float zy = vl.y + vr.y; zy = zy > 0.f ? zy : 0.2f * zy;
  float zz = vl.z + vr.z; zz = zz > 0.f ? zz : 0.2f * zz;
  float zw = vl.w + vr.w; zw = zw > 0.f ? zw : 0.2f * zw;
  float p = zx * va.x + zy * va.y + zz * va.z + zw * va.w;
#pragma unroll
  for (int off = 16; off > 0; off >>= 1) p += __shfl_xor(p, off, 32);
  if (lane == 0) {
    e[k] = p;
    atomicMax(&menc[d], f32_ord(p));
  }
}

// Pass B: ex = exp(e - m[dst]) (in place); segment sum via float atomics
__global__ void edge_exp(float* __restrict__ e, const int* __restrict__ dst,
                         const unsigned* __restrict__ menc,
                         float* __restrict__ sden, int E) {
  int k = blockIdx.x * blockDim.x + threadIdx.x;
  if (k >= E) return;
  int d = dst[k];
  float ex = __expf(e[k] - ord_f32(menc[d]));
  e[k] = ex;
  atomicAdd(&sden[d], ex);
}

// Pass C: acc[dst] += alpha * xl[src]  (128 f32 atomics per edge, 4 per lane)
__global__ void __launch_bounds__(256) edge_agg(
    const float* __restrict__ xl, const float* __restrict__ ex,
    const float* __restrict__ sden,
    const int* __restrict__ src, const int* __restrict__ dst,
    float* __restrict__ acc, int E) {
  int lane = threadIdx.x & 31;
  int k = blockIdx.x * 8 + (threadIdx.x >> 5);
  if (k >= E) return;
  int s = src[k], d = dst[k];
  float alpha = ex[k] / (sden[d] + 1e-16f);
  float4 vl = ((const float4*)(xl + (size_t)s * D))[lane];
  float* base = acc + (size_t)d * D + lane * 4;
  atomicAdd(base + 0, alpha * vl.x);
  atomicAdd(base + 1, alpha * vl.y);
  atomicAdd(base + 2, alpha * vl.z);
  atomicAdd(base + 3, alpha * vl.w);
}

// bias (+ optional relu)
__global__ void finalize(const float* __restrict__ acc, const float* __restrict__ bias,
                         float* __restrict__ out, int n, int relu) {
  int i = blockIdx.x * blockDim.x + threadIdx.x;
  if (i >= n) return;
  float v = acc[i] + bias[i & (D - 1)];
  if (relu) v = fmaxf(v, 0.f);
  out[i] = v;
}

// ---------------- driver ----------------
extern "C" void kernel_launch(void* const* d_in, const int* in_sizes, int n_in,
                              void* d_out, int out_size, void* d_ws, size_t ws_size,
                              hipStream_t stream) {
  (void)n_in; (void)out_size; (void)ws_size;
  const float* x    = (const float*)d_in[0];
  const float* Wl   = (const float*)d_in[1];
  const float* Wr   = (const float*)d_in[2];
  const float* att  = (const float*)d_in[3];
  const float* bias = (const float*)d_in[4];
  const int*   ei   = (const int*)d_in[5];

  const int N = in_sizes[0] / D;
  const int E = in_sizes[5] / 2;
  const int* src = ei;
  const int* dst = ei + E;

  // workspace layout (all offsets well aligned for these sizes)
  char* w = (char*)d_ws;
  size_t off = 0;
  float*   h    = (float*)(w + off);  off += (size_t)N * D * 4;
  __bf16*  hbf  = (__bf16*)(w + off); off += (size_t)N * D * 2;
  float*   xl   = (float*)(w + off);  off += (size_t)N * D * 4;
  float*   xr   = (float*)(w + off);  off += (size_t)N * D * 4;
  float*   e    = (float*)(w + off);  off += (size_t)E * 4;
  unsigned* menc = (unsigned*)(w + off); off += (size_t)N * 4;
  float*   sden = (float*)(w + off);  off += (size_t)N * 4;
  __bf16*  WlP  = (__bf16*)(w + off); off += (size_t)D * D * 2;
  __bf16*  WrP  = (__bf16*)(w + off); off += (size_t)D * D * 2;
  float*   acc  = (float*)d_out;      // scatter accumulator doubles as output

  const int rowPairs = (N + 31) / 32;     // 32 rows per wave

  for (int l = 0; l < 3; ++l) {
    const float* hin = (l == 0) ? x : h;

    cvt_f32_bf16<<<(N * D + 255) / 256, 256, 0, stream>>>(hin, hbf, N * D);
    pack_w<<<(D * D + 255) / 256, 256, 0, stream>>>(Wl + (size_t)l * D * D, WlP);
    pack_w<<<(D * D + 255) / 256, 256, 0, stream>>>(Wr + (size_t)l * D * D, WrP);

    gemm_bf16<<<dim3((rowPairs + 7) / 8, 2), 256, 0, stream>>>(hbf, WlP, WrP, xl, xr, N);

    hipMemsetAsync(acc,  0, (size_t)N * D * 4, stream);
    hipMemsetAsync(menc, 0, (size_t)N * 4, stream);
    hipMemsetAsync(sden, 0, (size_t)N * 4, stream);

    edge_score<<<(E + 7) / 8, 256, 0, stream>>>(xl, xr, src, dst, att + (size_t)l * D, e, menc, E);
    edge_exp<<<(E + 255) / 256, 256, 0, stream>>>(e, dst, menc, sden, E);
    edge_agg<<<(E + 7) / 8, 256, 0, stream>>>(xl, e, sden, src, dst, acc, E);

    finalize<<<(N * D + 255) / 256, 256, 0, stream>>>(
        acc, bias + (size_t)l * D, (l < 2) ? h : acc, N * D, (l < 2) ? 1 : 0);
  }
}
